// GVANet_45217415693012
// MI455X (gfx1250) — matrix operand order repaired and checked by reading, not compiled, not run on hardware
//
#include <hip/hip_runtime.h>

// ---------- types ----------
typedef _Float16 h8   __attribute__((ext_vector_type(8)));
typedef _Float16 v16h __attribute__((ext_vector_type(16)));
typedef float    v8f  __attribute__((ext_vector_type(8)));

#define BATCH 16
#define NPTS  2048
#define KNN_K 32
#define POS1  (BATCH * NPTS * 32)   // 1,048,576 (b,n,j) positions
#define NPOSE (BATCH * NPTS)        // 32,768 (b,n) positions

// ---------- WMMA fragment loaders (per CDNA5 ISA VGPR layouts) ----------
// A 16x32 f16 (MxK): lane L row M=L%16; halves 0..7 -> K = (L>=16?8:0)+i,
// halves 8..15 -> K = 16+(L>=16?8:0)+i  (two contiguous 16B chunks)
__device__ __forceinline__ v16h load_a_frag(const _Float16* wrow, int k0, int lane) {
  int ko = (lane >= 16) ? 8 : 0;
  h8 lo = *(const h8*)(wrow + k0 + ko);
  h8 hi = *(const h8*)(wrow + k0 + 16 + ko);
  v16h r;
#pragma unroll
  for (int i = 0; i < 8; ++i) { r[i] = lo[i]; r[i + 8] = hi[i]; }
  return r;
}
// B 32x16 f16 (KxN): lane L column N=L%16; halves 0..7 -> K = (L>=16?8:0)+i,
// halves 8..15 -> K = 16+(L>=16?8:0)+i  (two contiguous 16B chunks)
__device__ __forceinline__ v16h load_b_frag(const _Float16* xrow, int k0, int lane) {
  int ko = (lane >= 16) ? 8 : 0;
  h8 lo = *(const h8*)(xrow + k0 + ko);
  h8 hi = *(const h8*)(xrow + k0 + 16 + ko);
  v16h r;
#pragma unroll
  for (int i = 0; i < 8; ++i) { r[i] = lo[i]; r[i + 8] = hi[i]; }
  return r;
}

// ---------- universal WMMA GEMM with fused epilogue ----------
// Each wave owns one 16-position tile and MT consecutive 16-channel M-tiles,
// reusing each B (activation) fragment across the MT WMMAs -> 4x less
// streaming-side load traffic per WMMA vs one-tile-per-wave.
// Y[pos][o] = epilogue( sum_t sum_k W[t][o][k] * X[pos+t][k] )
// epilogue: relu_opt( (acc + pre[o]) * scale[o] + bias[o] )
// Yh: f16 out [pos][Opad]   |  Yf: f32 final out in (B, O, finalN) layout
template <int MT>
__global__ void __launch_bounds__(256) wmma_gemm_kernel(
    const _Float16* __restrict__ X, const _Float16* __restrict__ W,
    const float* __restrict__ pre, const float* __restrict__ scale,
    const float* __restrict__ bias, _Float16* __restrict__ Yh,
    float* __restrict__ Yf,
    int npos, int cinPad, int kw, int O, int Opad, int relu, int finalN,
    int mGroups) {
  int lane = threadIdx.x & 31;
  int wave = blockIdx.x * (blockDim.x >> 5) + (threadIdx.x >> 5);
  int posTiles = (npos + 15) >> 4;
  if (wave >= posTiles * mGroups) return;  // wave-uniform: EXEC stays all-ones
  int pt = wave % posTiles;
  int mtBase = (wave / posTiles) * MT;
  int p0 = pt << 4;
  int nn = lane & 15;

  v8f acc[MT];
#pragma unroll
  for (int u = 0; u < MT; ++u)
    acc[u] = (v8f){0.f, 0.f, 0.f, 0.f, 0.f, 0.f, 0.f, 0.f};

  for (int t = 0; t < kw; ++t) {
    int p = p0 + nn + t;
    if (p > npos - 1) p = npos - 1;           // clamp: only garbage slots affected
    const _Float16* xrow = X + (size_t)p * cinPad;
    const _Float16* wt0 =
        W + (size_t)t * Opad * cinPad + (size_t)((mtBase << 4) + nn) * cinPad;
    __builtin_prefetch(xrow + cinPad, 0, 3);  // next-row prefetch (global_prefetch_b8)
    for (int k0 = 0; k0 < cinPad; k0 += 32) {
      v16h b = load_b_frag(xrow, k0, lane);
#pragma unroll
      for (int u = 0; u < MT; ++u) {
        v16h a = load_a_frag(wt0 + (size_t)(u << 4) * cinPad, k0, lane);
        acc[u] = __builtin_amdgcn_wmma_f32_16x16x32_f16(false, a, false, b, (short)0,
                                                        acc[u], false, false);
      }
    }
  }

  // D 16x16 f32: lane L col N=L%16, VGPR v row M = v + (L>=16?8:0)
  int pp = p0 + nn;
  bool pvalid = pp < npos;
#pragma unroll
  for (int u = 0; u < MT; ++u) {
    int cbase = ((mtBase + u) << 4) + ((lane >> 4) << 3);
#pragma unroll
    for (int v = 0; v < 8; ++v) {
      int c = cbase + v;
      float val = acc[u][v];
      if (pre) val += (c < O) ? pre[c] : 0.f;
      if (scale) val = val * ((c < O) ? scale[c] : 0.f) + ((c < O) ? bias[c] : 0.f);
      if (relu) val = fmaxf(val, 0.f);
      if (Yh) {
        if (pvalid) Yh[(size_t)pp * Opad + c] = (_Float16)val;
      } else if (Yf) {
        if (pvalid && c < O) {
          int b_ = pp / finalN, n_ = pp - b_ * finalN;
          Yf[((size_t)b_ * O + c) * finalN + n_] = val;
        }
      }
    }
  }
}

// ---------- weight packing: (O,I,1,kw) f32 -> [t][Opad][Ipad] f16 ----------
__global__ void pack_w_kernel(_Float16* dst, const float* src, int O, int Opad,
                              int I, int Ipad, int kw) {
  int id = blockIdx.x * blockDim.x + threadIdx.x;
  int total = kw * Opad * Ipad;
  if (id >= total) return;
  int t = id / (Opad * Ipad);
  int r = id - t * Opad * Ipad;
  int o = r / Ipad;
  int i = r - o * Ipad;
  float v = 0.f;
  if (o < O && i < I) v = src[((size_t)o * I + i) * kw + t];
  dst[id] = (_Float16)v;
}

// ---------- kNN: C=3, x0 f32 [B][3][N] ----------
__global__ void __launch_bounds__(128) knn3_kernel(const float* __restrict__ x0,
                                                   int* __restrict__ idx) {
  __shared__ float chx[3][512];
  __shared__ float chn[512];
  __shared__ float td[128 * KNN_K];
  __shared__ int ti[128 * KNN_K];
  int b = blockIdx.y;
  int n = blockIdx.x * 128 + threadIdx.x;
  const float* xb = x0 + (size_t)b * 3 * NPTS;
  float q0 = xb[n], q1 = xb[NPTS + n], q2 = xb[2 * NPTS + n];
  float qq = q0 * q0 + q1 * q1 + q2 * q2;
  float* D = td + threadIdx.x * KNN_K;
  int* I = ti + threadIdx.x * KNN_K;
  int cnt = 0;
  for (int c0 = 0; c0 < NPTS; c0 += 512) {
    __syncthreads();
    for (int i = threadIdx.x; i < 512; i += 128) {
      int mm = c0 + i;
      float a = xb[mm], bb = xb[NPTS + mm], c = xb[2 * NPTS + mm];
      chx[0][i] = a; chx[1][i] = bb; chx[2][i] = c;
      chn[i] = a * a + bb * bb + c * c;
    }
    __syncthreads();
    for (int i = 0; i < 512; ++i) {
      int mm = c0 + i;
      float d = qq - 2.f * (q0 * chx[0][i] + q1 * chx[1][i] + q2 * chx[2][i]) + chn[i];
      if (cnt == KNN_K && d >= D[KNN_K - 1]) continue;
      int j = (cnt < KNN_K) ? cnt : KNN_K - 1;
      while (j > 0 && D[j - 1] > d) { D[j] = D[j - 1]; I[j] = I[j - 1]; --j; }
      D[j] = d; I[j] = mm;
      if (cnt < KNN_K) ++cnt;
    }
  }
  for (int k = 0; k < KNN_K; ++k) idx[((size_t)b * NPTS + n) * KNN_K + k] = I[k];
}

// ---------- kNN: C=32, x1 f16 [B*N][32] ----------
__global__ void __launch_bounds__(128) knn32_kernel(const _Float16* __restrict__ x1,
                                                    int* __restrict__ idx) {
  __shared__ float chx[128][33];
  __shared__ float chn[128];
  __shared__ float td[128 * KNN_K];
  __shared__ int ti[128 * KNN_K];
  int b = blockIdx.y;
  int n = blockIdx.x * 128 + threadIdx.x;
  const _Float16* xb = x1 + (size_t)b * NPTS * 32;
  float q[32];
  float qq = 0.f;
#pragma unroll
  for (int c = 0; c < 32; ++c) { q[c] = (float)xb[(size_t)n * 32 + c]; qq += q[c] * q[c]; }
  float* D = td + threadIdx.x * KNN_K;
  int* I = ti + threadIdx.x * KNN_K;
  int cnt = 0;
  for (int c0 = 0; c0 < NPTS; c0 += 128) {
    __syncthreads();
    {
      int i = threadIdx.x;
      int mm = c0 + i;
      float s = 0.f;
#pragma unroll
      for (int c = 0; c < 32; ++c) {
        float v = (float)xb[(size_t)mm * 32 + c];
        chx[i][c] = v;
        s += v * v;
      }
      chn[i] = s;
    }
    __syncthreads();
    for (int i = 0; i < 128; ++i) {
      int mm = c0 + i;
      float dot = 0.f;
#pragma unroll
      for (int c = 0; c < 32; ++c) dot += q[c] * chx[i][c];
      float d = qq - 2.f * dot + chn[i];
      if (cnt == KNN_K && d >= D[KNN_K - 1]) continue;
      int j = (cnt < KNN_K) ? cnt : KNN_K - 1;
      while (j > 0 && D[j - 1] > d) { D[j] = D[j - 1]; I[j] = I[j - 1]; --j; }
      D[j] = d; I[j] = mm;
      if (cnt < KNN_K) ++cnt;
    }
  }
  for (int k = 0; k < KNN_K; ++k) idx[((size_t)b * NPTS + n) * KNN_K + k] = I[k];
}

// ---------- edge feature gather + bn0 (C=3 -> 6ch padded to 32) ----------
__global__ void gather3_kernel(const float* __restrict__ x0, const int* __restrict__ idx,
                               const float* __restrict__ s0, const float* __restrict__ b0,
                               _Float16* __restrict__ F) {
  int pos = blockIdx.x * blockDim.x + threadIdx.x;
  if (pos >= POS1) return;
  int n = (pos >> 5) % NPTS;
  int b = pos / (32 * NPTS);
  int iv = idx[pos];
  const float* xb = x0 + (size_t)b * 3 * NPTS;
  _Float16* d = F + (size_t)pos * 32;
#pragma unroll
  for (int c = 0; c < 32; ++c) d[c] = (_Float16)0.f;
#pragma unroll
  for (int c = 0; c < 3; ++c) {
    float xc = xb[c * NPTS + n];
    float xf = xb[c * NPTS + iv];
    d[c]     = (_Float16)((xf - xc) * s0[c] + b0[c]);
    d[c + 3] = (_Float16)(xc * s0[c + 3] + b0[c + 3]);
  }
}

// ---------- edge feature gather + bn0 (C=32 -> 64ch) ----------
__global__ void gather32_kernel(const _Float16* __restrict__ x1, const int* __restrict__ idx,
                                const float* __restrict__ s0, const float* __restrict__ b0,
                                _Float16* __restrict__ F) {
  int pos = blockIdx.x * blockDim.x + threadIdx.x;
  if (pos >= POS1) return;
  int n = (pos >> 5) % NPTS;
  int b = pos / (32 * NPTS);
  int iv = idx[pos];
  const _Float16* xb = x1 + (size_t)b * NPTS * 32;
  _Float16* d = F + (size_t)pos * 64;
#pragma unroll
  for (int c = 0; c < 32; ++c) {
    float xc = (float)xb[(size_t)n * 32 + c];
    float xf = (float)xb[(size_t)iv * 32 + c];
    d[c]      = (_Float16)((xf - xc) * s0[c] + b0[c]);
    d[c + 32] = (_Float16)(xc * s0[c + 32] + b0[c + 32]);
  }
}

// ---------- max over j<J (per (b,n), per channel) ----------
__global__ void maxj_kernel(const _Float16* __restrict__ act, _Float16* __restrict__ out,
                            int NPp, int C, int J) {
  int id = blockIdx.x * blockDim.x + threadIdx.x;
  if (id >= NPp * C) return;
  int p = id / C, c = id - p * C;
  const _Float16* base = act + ((size_t)p * 32) * C + c;
  float m = (float)base[0];
  for (int j = 1; j < J; ++j) { float v = (float)base[(size_t)j * C]; m = v > m ? v : m; }
  out[(size_t)p * C + c] = (_Float16)m;
}

// ---------- concat x1(32) | x2(64) -> 96ch ----------
__global__ void concat96_kernel(const _Float16* __restrict__ x1, const _Float16* __restrict__ x2,
                                _Float16* __restrict__ m0) {
  int id = blockIdx.x * blockDim.x + threadIdx.x;
  if (id >= NPOSE * 96) return;
  int p = id / 96, c = id - p * 96;
  m0[id] = (c < 32) ? x1[(size_t)p * 32 + c] : x2[(size_t)p * 64 + (c - 32)];
}

// ---------- mean over N per (b, channel) ----------
__global__ void mean_kernel(const _Float16* __restrict__ X, float* __restrict__ mv) {
  int b = blockIdx.x, c = threadIdx.x;  // 256 channels
  float s = 0.f;
  for (int n = 0; n < NPTS; ++n) s += (float)X[((size_t)b * NPTS + n) * 256 + c];
  mv[b * 256 + c] = s / (float)NPTS;
}

// ---------- label input pack (16x16 f32 -> 16x32 f16 padded) ----------
__global__ void pack_lin_kernel(const float* __restrict__ l, _Float16* __restrict__ lin) {
  int id = blockIdx.x * blockDim.x + threadIdx.x;
  if (id >= 16 * 32) return;
  int r = id >> 5, c = id & 31;
  lin[id] = (_Float16)((c < 16) ? l[r * 16 + c] : 0.f);
}

// ---------- pack H = [mean(256) | lf(256) | x2(64)] per position ----------
__global__ void packH_kernel(const float* __restrict__ mv, const _Float16* __restrict__ lf,
                             const _Float16* __restrict__ x2, _Float16* __restrict__ H) {
  int id = blockIdx.x * blockDim.x + threadIdx.x;
  if (id >= NPOSE * 576) return;
  int p = id / 576, c = id - p * 576;
  int b = p / NPTS;
  float v;
  if (c < 256) v = mv[b * 256 + c];
  else if (c < 512) v = (float)lf[b * 256 + (c - 256)];
  else v = (float)x2[(size_t)p * 64 + (c - 512)];
  H[id] = (_Float16)v;
}

// ---------- host helpers ----------
static void gemm(const void* X, const void* W, const float* pre, const float* scale,
                 const float* bias, void* Yh, float* Yf, int npos, int cinPad, int kw,
                 int O, int Opad, int relu, int finalN, hipStream_t s) {
  int posTiles = (npos + 15) / 16;
  int mTiles = Opad / 16;
  int MT, mGroups;
  if (mTiles % 4 == 0) { MT = 4; mGroups = mTiles / 4; }
  else if (mTiles == 3) { MT = 3; mGroups = 1; }
  else { MT = 2; mGroups = 1; }
  long waves = (long)posTiles * mGroups;
  int blocks = (int)((waves + 7) / 8);
  if (MT == 4)
    wmma_gemm_kernel<4><<<blocks, 256, 0, s>>>((const _Float16*)X, (const _Float16*)W, pre,
                                               scale, bias, (_Float16*)Yh, Yf, npos, cinPad,
                                               kw, O, Opad, relu, finalN, mGroups);
  else if (MT == 3)
    wmma_gemm_kernel<3><<<blocks, 256, 0, s>>>((const _Float16*)X, (const _Float16*)W, pre,
                                               scale, bias, (_Float16*)Yh, Yf, npos, cinPad,
                                               kw, O, Opad, relu, finalN, mGroups);
  else
    wmma_gemm_kernel<2><<<blocks, 256, 0, s>>>((const _Float16*)X, (const _Float16*)W, pre,
                                               scale, bias, (_Float16*)Yh, Yf, npos, cinPad,
                                               kw, O, Opad, relu, finalN, mGroups);
}

extern "C" void kernel_launch(void* const* d_in, const int* in_sizes, int n_in,
                              void* d_out, int out_size, void* d_ws, size_t ws_size,
                              hipStream_t stream) {
  (void)in_sizes; (void)n_in; (void)out_size; (void)ws_size;
  char* ws = (char*)d_ws;
  size_t off = 0;
  auto alloc = [&](size_t bytes) -> void* {
    off = (off + 255) & ~(size_t)255;
    void* p = ws + off;
    off += bytes;
    return p;
  };

  // workspace
  int* idx1 = (int*)alloc((size_t)POS1 * 4);
  int* idx2 = (int*)alloc((size_t)POS1 * 4);
  void* regB = alloc((size_t)POS1 * 64 * 2 + 4096);  // ping
  void* regC = alloc((size_t)POS1 * 64 * 2 + 4096);  // pong
  void* x1 = alloc((size_t)NPOSE * 32 * 2);
  void* x2 = alloc((size_t)NPOSE * 64 * 2);
  void* m0 = alloc((size_t)NPOSE * 96 * 2);
  void* ma1 = alloc((size_t)NPOSE * 64 * 2);
  void* ma2 = alloc((size_t)NPOSE * 64 * 2);
  void* xf = alloc((size_t)NPOSE * 256 * 2);
  float* meanv = (float*)alloc(16 * 256 * 4);
  void* lin = alloc(16 * 32 * 2);
  void* la1 = alloc(16 * 32 * 2);
  void* la2 = alloc(16 * 64 * 2);
  void* lf = alloc(16 * 256 * 2);
  void* H = alloc((size_t)NPOSE * 576 * 2);
  void* ca1 = alloc((size_t)NPOSE * 128 * 2);
  void* ca2 = alloc((size_t)NPOSE * 64 * 2);
  void* wp11 = alloc(1 * 32 * 32 * 2);   void* wp12 = alloc(2 * 32 * 32 * 2);
  void* wp13 = alloc(4 * 32 * 32 * 2);   void* wp14 = alloc(8 * 32 * 32 * 2);
  void* wp21 = alloc(1 * 64 * 64 * 2);   void* wp22 = alloc(2 * 64 * 64 * 2);
  void* wp23 = alloc(4 * 64 * 64 * 2);   void* wp24 = alloc(8 * 64 * 64 * 2);
  void* wp31 = alloc(64 * 96 * 2);       void* wp32 = alloc(64 * 64 * 2);
  void* wp33 = alloc(256 * 64 * 2);
  void* wl1 = alloc(32 * 32 * 2);        void* wl2 = alloc(64 * 32 * 2);
  void* wl3 = alloc(256 * 64 * 2);
  void* wc1 = alloc((size_t)128 * 576 * 2); void* wc2 = alloc(64 * 128 * 2);
  void* wc3 = alloc(48 * 64 * 2);

  auto F32 = [&](int i) { return (const float*)d_in[i]; };
  auto packw = [&](void* dst, int si, int O, int Opad, int I, int Ipad, int kw) {
    int total = kw * Opad * Ipad;
    pack_w_kernel<<<(total + 255) / 256, 256, 0, stream>>>((_Float16*)dst, F32(si), O, Opad, I,
                                                           Ipad, kw);
  };

  // ---- weight packing (input leaf indices per setup_inputs tree order) ----
  packw(wp11, 4, 32, 32, 6, 32, 1);    packw(wp12, 7, 32, 32, 32, 32, 2);
  packw(wp13, 10, 32, 32, 32, 32, 4);  packw(wp14, 13, 32, 32, 32, 32, 8);
  packw(wp21, 18, 64, 64, 64, 64, 1);  packw(wp22, 21, 64, 64, 64, 64, 2);
  packw(wp23, 24, 64, 64, 64, 64, 4);  packw(wp24, 27, 64, 64, 64, 64, 8);
  packw(wp31, 30, 64, 64, 96, 96, 1);  packw(wp32, 33, 64, 64, 64, 64, 1);
  packw(wp33, 36, 256, 256, 64, 64, 1);
  packw(wl1, 39, 32, 32, 16, 32, 1);   packw(wl2, 42, 64, 64, 32, 32, 1);
  packw(wl3, 45, 256, 256, 64, 64, 1);
  packw(wc1, 48, 128, 128, 576, 576, 1); packw(wc2, 52, 64, 64, 128, 128, 1);
  packw(wc3, 56, 40, 48, 64, 64, 1);

  // ---- stage 1: knn(x0), edge features, conv1 chain, max ----
  knn3_kernel<<<dim3(NPTS / 128, BATCH), 128, 0, stream>>>(F32(0), idx1);
  gather3_kernel<<<(POS1 + 255) / 256, 256, 0, stream>>>(F32(0), idx1, F32(2), F32(3),
                                                         (_Float16*)regB);
  gemm(regB, wp11, nullptr, F32(5), F32(6), regC, nullptr, POS1, 32, 1, 32, 32, 1, 0, stream);
  gemm(regC, wp12, nullptr, F32(8), F32(9), regB, nullptr, POS1, 32, 2, 32, 32, 1, 0, stream);
  gemm(regB, wp13, nullptr, F32(11), F32(12), regC, nullptr, POS1, 32, 4, 32, 32, 1, 0, stream);
  gemm(regC, wp14, nullptr, F32(14), F32(15), regB, nullptr, POS1, 32, 8, 32, 32, 1, 0, stream);
  maxj_kernel<<<(NPOSE * 32 + 255) / 256, 256, 0, stream>>>((const _Float16*)regB,
                                                            (_Float16*)x1, NPOSE, 32, 21);

  // ---- stage 2: knn(x1), edge features, conv2 chain, max ----
  knn32_kernel<<<dim3(NPTS / 128, BATCH), 128, 0, stream>>>((const _Float16*)x1, idx2);
  gather32_kernel<<<(POS1 + 255) / 256, 256, 0, stream>>>((const _Float16*)x1, idx2, F32(16),
                                                          F32(17), (_Float16*)regB);
  gemm(regB, wp21, nullptr, F32(19), F32(20), regC, nullptr, POS1, 64, 1, 64, 64, 1, 0, stream);
  gemm(regC, wp22, nullptr, F32(22), F32(23), regB, nullptr, POS1, 64, 2, 64, 64, 1, 0, stream);
  gemm(regB, wp23, nullptr, F32(25), F32(26), regC, nullptr, POS1, 64, 4, 64, 64, 1, 0, stream);
  gemm(regC, wp24, nullptr, F32(28), F32(29), regB, nullptr, POS1, 64, 8, 64, 64, 1, 0, stream);
  maxj_kernel<<<(NPOSE * 64 + 255) / 256, 256, 0, stream>>>((const _Float16*)regB,
                                                            (_Float16*)x2, NPOSE, 64, 21);

  // ---- stage 3: mlp1d on concat(x1,x2) ----
  concat96_kernel<<<(NPOSE * 96 + 255) / 256, 256, 0, stream>>>((const _Float16*)x1,
                                                                (const _Float16*)x2,
                                                                (_Float16*)m0);
  gemm(m0, wp31, nullptr, F32(31), F32(32), ma1, nullptr, NPOSE, 96, 1, 64, 64, 1, 0, stream);
  gemm(ma1, wp32, nullptr, F32(34), F32(35), ma2, nullptr, NPOSE, 64, 1, 64, 64, 1, 0, stream);
  gemm(ma2, wp33, nullptr, F32(37), F32(38), xf, nullptr, NPOSE, 64, 1, 256, 256, 1, 0, stream);
  mean_kernel<<<BATCH, 256, 0, stream>>>((const _Float16*)xf, meanv);

  // ---- stage 4: label mlp ----
  pack_lin_kernel<<<2, 256, 0, stream>>>(F32(1), (_Float16*)lin);
  gemm(lin, wl1, nullptr, F32(40), F32(41), la1, nullptr, 16, 32, 1, 32, 32, 1, 0, stream);
  gemm(la1, wl2, nullptr, F32(43), F32(44), la2, nullptr, 16, 32, 1, 64, 64, 1, 0, stream);
  gemm(la2, wl3, nullptr, F32(46), F32(47), lf, nullptr, 16, 64, 1, 256, 256, 1, 0, stream);

  // ---- stage 5: classification head ----
  packH_kernel<<<(NPOSE * 576 + 255) / 256, 256, 0, stream>>>(meanv, (const _Float16*)lf,
                                                              (const _Float16*)x2, (_Float16*)H);
  gemm(H, wc1, F32(49), F32(50), F32(51), ca1, nullptr, NPOSE, 576, 1, 128, 128, 1, 0, stream);
  gemm(ca1, wc2, F32(53), F32(54), F32(55), ca2, nullptr, NPOSE, 128, 1, 64, 64, 1, 0, stream);
  gemm(ca2, wc3, F32(57), nullptr, nullptr, nullptr, (float*)d_out, NPOSE, 64, 1, 40, 48, 0,
       NPTS, stream);
}
